// SpectralMamaba_66236985639734
// MI455X (gfx1250) — matrix operand order repaired and checked by reading, not compiled, not run on hardware
//
#include <hip/hip_runtime.h>
#include <hip/hip_bf16.h>

// ---------------------------------------------------------------------------
// SpectralMamba block for MI455X (gfx1250, wave32).
// - GEMMs: v_wmma_f32_16x16x32_bf16, one 16x64 output strip per wave
//   (4 f32 accumulators; A loaded once per K-step, 4 WMMAs per A load).
// - Weight panels staged block-wide into LDS via global_load_async_to_lds_b128
//   (ASYNCcnt) + s_wait_asynccnt, B-tiles then read from LDS.
// - Selective scan: register-resident sequential kernel (1 block per (b,k)).
// ---------------------------------------------------------------------------

typedef __attribute__((ext_vector_type(16))) __bf16 v16bf;
typedef __attribute__((ext_vector_type(8)))  float  v8f;

#define L_  4096
#define DI_ 128

// ---- WMMA helpers ---------------------------------------------------------

__device__ __forceinline__ v16bf load_a_row(const __bf16* p) {
  // 16-bit A layout: lane<16 -> row M=lane, K = [0..7] and [16..23];
  // lane>=16 -> same row, K = [8..15] and [24..31]. Caller pre-offsets p by
  // ((lane>>4)<<3) so we always read [0..7] and [16..23] relative to p.
  v16bf a;
#pragma unroll
  for (int i = 0; i < 8; ++i) { a[i] = p[i]; a[i + 8] = p[i + 16]; }
  return a;
}

__device__ __forceinline__ v16bf load_a_tile(const __bf16* tile, int lda, int lane) {
  const __bf16* p = tile + (size_t)(lane & 15) * lda + ((lane >> 4) << 3);
  return load_a_row(p);
}

__device__ __forceinline__ v16bf load_b_tile(const __bf16* wt, int ldb, int lane) {
  // B 32x16 layout: lanes 0-15 hold K=[0..7],[16..23] of column N=lane,
  // lanes 16-31 hold K=[8..15],[24..31]. Weights stored out-major [N][K].
  const __bf16* p = wt + (size_t)(lane & 15) * ldb + ((lane >> 4) << 3);
  v16bf b;
#pragma unroll
  for (int i = 0; i < 8; ++i) { b[i] = p[i]; b[i + 8] = p[i + 16]; }
  return b;
}

__device__ __forceinline__ v8f wmma_bf16(v16bf a, v16bf b, v8f c) {
  return __builtin_amdgcn_wmma_f32_16x16x32_bf16(false, a, false, b,
                                                 (short)0, c, false, false);
}

// Async-stage `n16` 16-byte chunks of weights into LDS offset 0 (the staging
// kernels declare exactly one __shared__ array, so it sits at LDS offset 0).
// Tracked by ASYNCcnt; waits on it and barriers before returning.
__device__ __forceinline__ void stage_lds_async(const __bf16* W, int n16) {
  for (int i = threadIdx.x; i < n16; i += 256) {
    unsigned laddr = (unsigned)i * 16u;
    unsigned long long gaddr = (unsigned long long)(const void*)(W + i * 8);
    asm volatile("global_load_async_to_lds_b128 %0, %1, off"
                 :: "v"(laddr), "v"(gaddr) : "memory");
  }
  asm volatile("s_wait_asynccnt 0x0" ::: "memory");
  __syncthreads();
}

// scan-direction pixel index map (transpose / reverse are involutions)
__device__ __forceinline__ int mapdir(int k, int l) {
  int t = ((l & 63) << 6) | (l >> 6);
  switch (k & 3) {
    case 0:  return l;
    case 1:  return t;
    case 2:  return 4095 - l;
    default: return 4095 - t;
  }
}

// ---- weight prep: fp32 -> bf16 (x_proj padded 36 -> 48 cols) --------------

__global__ void prep_weights(const float* __restrict__ pw1,
                             const float* __restrict__ pw2,
                             const float* __restrict__ inp,
                             const float* __restrict__ outp,
                             const float* __restrict__ xpw,
                             __bf16* __restrict__ wbf) {
  int e = blockIdx.x * blockDim.x + threadIdx.x;
  if (e >= 57344) return;
  float v;
  if (e < 4096)        v = pw1[e];
  else if (e < 8192)   v = pw2[e - 4096];
  else if (e < 24576)  v = inp[e - 8192];
  else if (e < 32768)  v = outp[e - 24576];
  else {
    int i = e - 32768;
    int k = i / 6144, rem = i - k * 6144;
    int c = rem >> 7, d = rem & 127;
    v = (c < 36) ? xpw[(k * 36 + c) * 128 + d] : 0.0f;
  }
  wbf[e] = (__bf16)v;
}

// ---- score path -----------------------------------------------------------

__global__ void score_max(const float* __restrict__ x, float* __restrict__ s0) {
  __shared__ float sm[256];
  int bc = blockIdx.x;                       // b*64+c
  const float* xp = x + (size_t)bc * L_;
  float m = -3.4e38f;
  for (int i = threadIdx.x; i < L_; i += 256) m = fmaxf(m, xp[i]);
  sm[threadIdx.x] = m; __syncthreads();
  for (int st = 128; st; st >>= 1) {
    if (threadIdx.x < st) sm[threadIdx.x] = fmaxf(sm[threadIdx.x], sm[threadIdx.x + st]);
    __syncthreads();
  }
  if (threadIdx.x == 0) s0[bc] = sm[0];
}

__global__ void score_mlp(const float* __restrict__ s0,
                          const float* __restrict__ w1, const float* __restrict__ b1,
                          const float* __restrict__ w2, const float* __restrict__ b2,
                          float* __restrict__ sc) {
  __shared__ float sh[64], sh2[64];
  int b = blockIdx.x, t = threadIdx.x;       // 64 threads
  sh[t] = s0[b * 64 + t];
  __syncthreads();
  float a = b1[t];
  for (int i = 0; i < 64; ++i) a += w1[t * 64 + i] * sh[i];
  sh2[t] = fmaxf(a, 0.0f);
  __syncthreads();
  float a2 = b2[t];
  for (int i = 0; i < 64; ++i) a2 += w2[t * 64 + i] * sh2[i];
  sc[b * 64 + t] = fmaxf(a2, 0.0f);
}

// ---- depthwise 3x3 convs --------------------------------------------------

__global__ void dwconv1_kernel(const float* __restrict__ x, const float* __restrict__ w,
                               const float* __restrict__ bias, __bf16* __restrict__ t1) {
  int e = blockIdx.x * blockDim.x + threadIdx.x;          // NHWC output index
  int c = e & 63, l = (e >> 6) & 4095, b = e >> 18;
  int hh = l >> 6, ww = l & 63;
  const float* xp = x + ((size_t)(b * 64 + c)) * L_;      // NCHW input
  const float* wp = w + c * 9;
  float s = bias[c];
#pragma unroll
  for (int dy = -1; dy <= 1; ++dy)
#pragma unroll
    for (int dx = -1; dx <= 1; ++dx) {
      int yy = hh + dy, xx = ww + dx;
      if (yy >= 0 && yy < 64 && xx >= 0 && xx < 64)
        s += xp[yy * 64 + xx] * wp[(dy + 1) * 3 + (dx + 1)];
    }
  t1[e] = (__bf16)s;
}

__global__ void conv128_kernel(const float* __restrict__ xi, const float* __restrict__ w,
                               const float* __restrict__ bias, __bf16* __restrict__ xc) {
  int e = blockIdx.x * blockDim.x + threadIdx.x;          // NHWC output, 128 ch
  int c = e & 127, l = (e >> 7) & 4095, b = e >> 19;
  int hh = l >> 6, ww = l & 63;
  const float* xp = xi + ((size_t)(b * 128 + c)) * L_;    // NCHW input
  const float* wp = w + c * 9;
  float s = bias[c];
#pragma unroll
  for (int dy = -1; dy <= 1; ++dy)
#pragma unroll
    for (int dx = -1; dx <= 1; ++dx) {
      int yy = hh + dy, xx = ww + dx;
      if (yy >= 0 && yy < 64 && xx >= 0 && xx < 64)
        s += xp[yy * 64 + xx] * wp[(dy + 1) * 3 + (dx + 1)];
    }
  s = s / (1.0f + __expf(-s));                            // SiLU
  xc[e] = (__bf16)s;
}

__global__ void dwconv2_kernel(const float* __restrict__ hsin, const float* __restrict__ w,
                               const float* __restrict__ bias, __bf16* __restrict__ t2) {
  int e = blockIdx.x * blockDim.x + threadIdx.x;          // NHWC in and out
  int c = e & 63, l = (e >> 6) & 4095, b = e >> 18;
  int hh = l >> 6, ww = l & 63;
  const float* wp = w + c * 9;
  float s = bias[c];
#pragma unroll
  for (int dy = -1; dy <= 1; ++dy)
#pragma unroll
    for (int dx = -1; dx <= 1; ++dx) {
      int yy = hh + dy, xx = ww + dx;
      if (yy >= 0 && yy < 64 && xx >= 0 && xx < 64)
        s += hsin[((size_t)b * L_ + yy * 64 + xx) * 64 + c] * wp[(dy + 1) * 3 + (dx + 1)];
    }
  t2[e] = (__bf16)s;
}

// ---- batch mean/var (deterministic two-stage) -----------------------------

__global__ void bstat_partial(const float* __restrict__ h,
                              float* __restrict__ psum, float* __restrict__ psq) {
  __shared__ float ss[256], sq[256];
  int blk = blockIdx.x;                      // b*256 + chunk
  int b = blk >> 8, chunk = blk & 255;
  size_t base = (size_t)b * 262144 + (size_t)chunk * 1024;
  float s = 0.0f, q = 0.0f;
  for (int i = threadIdx.x; i < 1024; i += 256) {
    float v = h[base + i]; s += v; q += v * v;
  }
  ss[threadIdx.x] = s; sq[threadIdx.x] = q; __syncthreads();
  for (int st = 128; st; st >>= 1) {
    if (threadIdx.x < st) { ss[threadIdx.x] += ss[threadIdx.x + st];
                            sq[threadIdx.x] += sq[threadIdx.x + st]; }
    __syncthreads();
  }
  if (threadIdx.x == 0) { psum[blk] = ss[0]; psq[blk] = sq[0]; }
}

__global__ void bstat_final(const float* __restrict__ psum, const float* __restrict__ psq,
                            float* __restrict__ stats) {
  __shared__ float ss[256], sq[256];
  int b = blockIdx.x;
  ss[threadIdx.x] = psum[b * 256 + threadIdx.x];
  sq[threadIdx.x] = psq[b * 256 + threadIdx.x];
  __syncthreads();
  for (int st = 128; st; st >>= 1) {
    if (threadIdx.x < st) { ss[threadIdx.x] += ss[threadIdx.x + st];
                            sq[threadIdx.x] += sq[threadIdx.x + st]; }
    __syncthreads();
  }
  if (threadIdx.x == 0) {
    float m = ss[0] * (1.0f / 262144.0f);
    float var = sq[0] * (1.0f / 262144.0f) - m * m;
    stats[b * 2] = m; stats[b * 2 + 1] = rsqrtf(var + 1e-5f);
  }
}

// ---- batch-norm + channel LN (per-pixel, one wave per pixel) --------------

__global__ void bn_ln_kernel(const float* __restrict__ h, const float* __restrict__ stats,
                             const float* __restrict__ g, const float* __restrict__ bta,
                             float* __restrict__ xhbn, __bf16* __restrict__ xhln) {
  int lane = threadIdx.x & 31;
  int wid = blockIdx.x * 8 + (threadIdx.x >> 5);   // pixel = b*L + l
  int b = wid >> 12;
  float mB = stats[b * 2], iB = stats[b * 2 + 1];
  const float* hp = h + (size_t)wid * 64;
  float v0 = (hp[lane] - mB) * iB;
  float v1 = (hp[lane + 32] - mB) * iB;
  xhbn[(size_t)wid * 64 + lane]      = v0;
  xhbn[(size_t)wid * 64 + lane + 32] = v1;
  float s = v0 + v1, q = v0 * v0 + v1 * v1;
  for (int off = 16; off; off >>= 1) { s += __shfl_xor(s, off); q += __shfl_xor(q, off); }
  float m = s * (1.0f / 64.0f);
  float var = q * (1.0f / 64.0f) - m * m;
  float rinv = rsqrtf(var + 1e-5f);
  xhln[(size_t)wid * 64 + lane]      = (__bf16)((v0 - m) * rinv * g[lane] + bta[lane]);
  xhln[(size_t)wid * 64 + lane + 32] = (__bf16)((v1 - m) * rinv * g[lane + 32] + bta[lane + 32]);
}

// ---- WMMA GEMM kernels (16x64 strip per wave, LDS-staged weights) ---------

// pw1: [16384x64] x [64x64] + bias, ReLU -> h NHWC f32.  grid 128x256
__global__ void pw1_gemm(const __bf16* __restrict__ A, const __bf16* __restrict__ W,
                         const float* __restrict__ bias, float* __restrict__ out) {
  __shared__ __bf16 wlds[64 * 64];                 // 8 KB
  stage_lds_async(W, 512);
  int lane = threadIdx.x & 31;
  int wid = blockIdx.x * 8 + (threadIdx.x >> 5);   // row tile, 1024 total
  int row0 = wid * 16;
  v8f acc[4] = {};
#pragma unroll
  for (int k0 = 0; k0 < 64; k0 += 32) {
    v16bf a = load_a_tile(A + (size_t)row0 * 64 + k0, 64, lane);
#pragma unroll
    for (int j = 0; j < 4; ++j) {
      v16bf b = load_b_tile(wlds + j * 16 * 64 + k0, 64, lane);
      acc[j] = wmma_bf16(a, b, acc[j]);
    }
  }
  int rbase = row0 + ((lane >> 4) << 3);
#pragma unroll
  for (int j = 0; j < 4; ++j) {
    int col = j * 16 + (lane & 15);
    float bv = bias[col];
#pragma unroll
    for (int r = 0; r < 8; ++r)
      out[(size_t)(rbase + r) * 64 + col] = fmaxf(acc[j][r] + bv, 0.0f);
  }
}

// in_proj: [16384x64] x [64x256] + bias -> xi (NCHW, cols<128), z (NHWC).
// Each wave does a 64-col group (tq). grid 512x256
__global__ void inproj_gemm(const __bf16* __restrict__ A, const __bf16* __restrict__ W,
                            const float* __restrict__ bias,
                            float* __restrict__ xi, float* __restrict__ z) {
  __shared__ __bf16 wlds[256 * 64];                // 32 KB
  stage_lds_async(W, 2048);
  int lane = threadIdx.x & 31;
  int wid = blockIdx.x * 8 + (threadIdx.x >> 5);   // 4096 waves
  int tm = wid >> 2, tq = wid & 3;
  int row0 = tm * 16;
  v8f acc[4] = {};
#pragma unroll
  for (int k0 = 0; k0 < 64; k0 += 32) {
    v16bf a = load_a_tile(A + (size_t)row0 * 64 + k0, 64, lane);
#pragma unroll
    for (int j = 0; j < 4; ++j) {
      v16bf b = load_b_tile(wlds + (tq * 64 + j * 16) * 64 + k0, 64, lane);
      acc[j] = wmma_bf16(a, b, acc[j]);
    }
  }
  int rbase = row0 + ((lane >> 4) << 3);
#pragma unroll
  for (int j = 0; j < 4; ++j) {
    int col = tq * 64 + j * 16 + (lane & 15);
    float bv = bias[col];
#pragma unroll
    for (int r = 0; r < 8; ++r) {
      int row = rbase + r;
      float v = acc[j][r] + bv;
      int bb = row >> 12, l = row & 4095;
      if (col < 128) xi[((size_t)(bb * 128 + col)) * L_ + l] = v;   // NCHW for dwconv
      else           z[(size_t)row * 128 + (col - 128)] = v;        // NHWC
    }
  }
}

// x_proj: per (b,k): [4096x128] x [128x48pad] -> x_dbl[b,k,l,48].
// wid = k*1024 + tm; a block's 8 waves share k, stage that k's 12 KB panel.
// grid 512x256
__global__ void xproj_gemm(const __bf16* __restrict__ xc, const __bf16* __restrict__ Wp,
                           float* __restrict__ xdbl) {
  __shared__ __bf16 wlds[48 * 128];                // 12 KB
  int kblk = blockIdx.x >> 7;                      // k shared by whole block
  stage_lds_async(Wp + (size_t)kblk * 48 * 128, 768);
  int lane = threadIdx.x & 31;
  int wid = blockIdx.x * 8 + (threadIdx.x >> 5);
  int k = wid >> 10, tm = wid & 1023;
  int row0 = tm * 16;
  int b = row0 >> 12, l0 = row0 & 4095;
  int ls = mapdir(k, l0 + (lane & 15));            // direction-permuted source pixel
  int ko8 = (lane >> 4) << 3;
  v8f acc[3] = {};
#pragma unroll
  for (int kc = 0; kc < 128; kc += 32) {
    v16bf a = load_a_row(xc + ((size_t)(b * L_ + ls)) * 128 + kc + ko8);
#pragma unroll
    for (int j = 0; j < 3; ++j) {
      v16bf bm = load_b_tile(wlds + (j * 16) * 128 + kc, 128, lane);
      acc[j] = wmma_bf16(a, bm, acc[j]);
    }
  }
  int lbase = l0 + ((lane >> 4) << 3);
#pragma unroll
  for (int j = 0; j < 3; ++j) {
    int col = j * 16 + (lane & 15);
#pragma unroll
    for (int r = 0; r < 8; ++r)
      xdbl[(((size_t)(b * 4 + k)) * L_ + (lbase + r)) * 48 + col] = acc[j][r];
  }
}

// out_proj: [16384x128] x [128x64] + bias, + xh residual, * score -> hs NHWC.
// grid 128x256
__global__ void outproj_gemm(const __bf16* __restrict__ A, const __bf16* __restrict__ W,
                             const float* __restrict__ bias, const float* __restrict__ xhbn,
                             const float* __restrict__ score, float* __restrict__ hs) {
  __shared__ __bf16 wlds[64 * 128];                // 16 KB
  stage_lds_async(W, 1024);
  int lane = threadIdx.x & 31;
  int wid = blockIdx.x * 8 + (threadIdx.x >> 5);   // row tile, 1024 total
  int row0 = wid * 16;
  v8f acc[4] = {};
#pragma unroll
  for (int kc = 0; kc < 128; kc += 32) {
    v16bf a = load_a_tile(A + (size_t)row0 * 128 + kc, 128, lane);
#pragma unroll
    for (int j = 0; j < 4; ++j) {
      v16bf b = load_b_tile(wlds + (j * 16) * 128 + kc, 128, lane);
      acc[j] = wmma_bf16(a, b, acc[j]);
    }
  }
  int rbase = row0 + ((lane >> 4) << 3);
#pragma unroll
  for (int j = 0; j < 4; ++j) {
    int col = j * 16 + (lane & 15);
    float bv = bias[col];
#pragma unroll
    for (int r = 0; r < 8; ++r) {
      int row = rbase + r;
      int bb = row >> 12;
      float v = acc[j][r] + bv + xhbn[(size_t)row * 64 + col];
      v *= score[bb * 64 + col];
      hs[(size_t)row * 64 + col] = v;
    }
  }
}

// pw2: [16384x64] x [64x64] + bias, ReLU, + input residual -> out NCHW.
// grid 128x256
__global__ void pw2_gemm(const __bf16* __restrict__ A, const __bf16* __restrict__ W,
                         const float* __restrict__ bias, const float* __restrict__ xres,
                         float* __restrict__ out) {
  __shared__ __bf16 wlds[64 * 64];                 // 8 KB
  stage_lds_async(W, 512);
  int lane = threadIdx.x & 31;
  int wid = blockIdx.x * 8 + (threadIdx.x >> 5);
  int row0 = wid * 16;
  v8f acc[4] = {};
#pragma unroll
  for (int k0 = 0; k0 < 64; k0 += 32) {
    v16bf a = load_a_tile(A + (size_t)row0 * 64 + k0, 64, lane);
#pragma unroll
    for (int j = 0; j < 4; ++j) {
      v16bf b = load_b_tile(wlds + j * 16 * 64 + k0, 64, lane);
      acc[j] = wmma_bf16(a, b, acc[j]);
    }
  }
  int rbase = row0 + ((lane >> 4) << 3);
#pragma unroll
  for (int j = 0; j < 4; ++j) {
    int col = j * 16 + (lane & 15);
    float bv = bias[col];
#pragma unroll
    for (int r = 0; r < 8; ++r) {
      int row = rbase + r;
      int bb = row >> 12, hw = row & 4095;
      float v = fmaxf(acc[j][r] + bv, 0.0f) + xres[((size_t)(bb * 64 + col)) * L_ + hw];
      out[((size_t)(bb * 64 + col)) * L_ + hw] = v;
    }
  }
}

// ---- selective scan: one block per (b,k), 128 threads (d), state in regs --

__global__ void scan_kernel(const float* __restrict__ xdbl, const __bf16* __restrict__ xc,
                            const float* __restrict__ dtw, const float* __restrict__ dtb,
                            const float* __restrict__ alog, const float* __restrict__ dskip,
                            float* __restrict__ ys) {
  int bk = blockIdx.x;                 // 0..15
  int b = bk >> 2, k = bk & 3;
  int d = threadIdx.x;                 // 0..127
  const float* wrow = dtw + (size_t)(k * DI_ + d) * 4;
  float w0 = wrow[0], w1 = wrow[1], w2 = wrow[2], w3 = wrow[3];
  float bb = dtb[k * DI_ + d];
  float Dd = dskip[k * DI_ + d];
  float Av[16];
#pragma unroll
  for (int n = 0; n < 16; ++n) Av[n] = -__expf(alog[(size_t)(k * DI_ + d) * 16 + n]);
  float st[16];
#pragma unroll
  for (int n = 0; n < 16; ++n) st[n] = 0.0f;

  __shared__ float sb[36];
  const float* xrow = xdbl + (size_t)bk * L_ * 48;
  float* yrow = ys + (size_t)bk * L_ * DI_;

  for (int l = 0; l < L_; ++l) {
    __syncthreads();
    if (d < 36) sb[d] = xrow[(size_t)l * 48 + d];
    __syncthreads();
    float u = w0 * sb[0] + w1 * sb[1] + w2 * sb[2] + w3 * sb[3] + bb;
    float delta = (u > 20.0f) ? u : log1pf(__expf(u));   // softplus
    int ls = mapdir(k, l);
    float xv = (float)xc[((size_t)(b * L_ + ls)) * DI_ + d];
    float dx = delta * xv;
    float y = Dd * xv;
#pragma unroll
    for (int n = 0; n < 16; ++n) {
      st[n] = __expf(delta * Av[n]) * st[n] + dx * sb[4 + n];
      y += st[n] * sb[20 + n];
    }
    yrow[(size_t)l * DI_ + d] = y;
  }
}

// ---- combine 4 directions + out-norm LN + SiLU(z) gate -> g bf16 NHWC -----

__global__ void combine_kernel(const float* __restrict__ ys, const float* __restrict__ z,
                               const float* __restrict__ g, const float* __restrict__ bta,
                               __bf16* __restrict__ gout) {
  int lane = threadIdx.x & 31;
  int wid = blockIdx.x * 8 + (threadIdx.x >> 5);   // pixel
  int b = wid >> 12, l = wid & 4095;
  int t = ((l & 63) << 6) | (l >> 6);
  int src[4] = { l, t, 4095 - l, 4095 - t };
  float v[4];
  float s = 0.0f, q = 0.0f;
#pragma unroll
  for (int j = 0; j < 4; ++j) {
    int d = lane + 32 * j;
    float acc = 0.0f;
#pragma unroll
    for (int k = 0; k < 4; ++k)
      acc += ys[(((size_t)(b * 4 + k)) * L_ + src[k]) * DI_ + d];
    v[j] = acc; s += acc; q += acc * acc;
  }
  for (int off = 16; off; off >>= 1) { s += __shfl_xor(s, off); q += __shfl_xor(q, off); }
  float m = s * (1.0f / 128.0f);
  float var = q * (1.0f / 128.0f) - m * m;
  float rinv = rsqrtf(var + 1e-5f);
#pragma unroll
  for (int j = 0; j < 4; ++j) {
    int d = lane + 32 * j;
    float zz = z[((size_t)(b * L_ + l)) * DI_ + d];
    float sil = zz / (1.0f + __expf(-zz));
    float val = ((v[j] - m) * rinv * g[d] + bta[d]) * sil;
    gout[((size_t)(b * L_ + l)) * DI_ + d] = (__bf16)val;
  }
}

// ---------------------------------------------------------------------------

extern "C" void kernel_launch(void* const* d_in, const int* in_sizes, int n_in,
                              void* d_out, int out_size, void* d_ws, size_t ws_size,
                              hipStream_t stream) {
  (void)in_sizes; (void)n_in; (void)out_size; (void)ws_size;
  const float* x          = (const float*)d_in[0];
  const float* lin1_w     = (const float*)d_in[1];
  const float* lin1_b     = (const float*)d_in[2];
  const float* lin2_w     = (const float*)d_in[3];
  const float* lin2_b     = (const float*)d_in[4];
  const float* dw1_w      = (const float*)d_in[5];
  const float* dw1_b      = (const float*)d_in[6];
  const float* pw1_w      = (const float*)d_in[7];
  const float* pw1_b      = (const float*)d_in[8];
  const float* dw2_w      = (const float*)d_in[9];
  const float* dw2_b      = (const float*)d_in[10];
  const float* pw2_w      = (const float*)d_in[11];
  const float* pw2_b      = (const float*)d_in[12];
  const float* ln_g       = (const float*)d_in[13];
  const float* ln_b       = (const float*)d_in[14];
  const float* in_proj_w  = (const float*)d_in[15];
  const float* in_proj_b  = (const float*)d_in[16];
  const float* conv_w     = (const float*)d_in[17];
  const float* conv_b     = (const float*)d_in[18];
  const float* x_proj_w   = (const float*)d_in[19];
  const float* dt_proj_w  = (const float*)d_in[20];
  const float* dt_proj_b  = (const float*)d_in[21];
  const float* A_log      = (const float*)d_in[22];
  const float* Ds         = (const float*)d_in[23];
  const float* out_norm_g = (const float*)d_in[24];
  const float* out_norm_b = (const float*)d_in[25];
  const float* out_proj_w = (const float*)d_in[26];
  const float* out_proj_b = (const float*)d_in[27];
  float* out = (float*)d_out;

  char* wsB = (char*)d_ws;
  size_t off = 0;
  auto alloc = [&](size_t bytes) {
    void* p = wsB + off;
    off = (off + bytes + 255) & ~(size_t)255;
    return p;
  };
  __bf16* wbf   = (__bf16*)alloc((size_t)57344 * 2);
  float*  score0= (float*) alloc((size_t)256 * 4);
  float*  score = (float*) alloc((size_t)256 * 4);
  float*  psum  = (float*) alloc((size_t)1024 * 4);
  float*  psq   = (float*) alloc((size_t)1024 * 4);
  float*  stats = (float*) alloc((size_t)8 * 4);
  __bf16* t1    = (__bf16*)alloc((size_t)1048576 * 2);
  float*  hbuf  = (float*) alloc((size_t)1048576 * 4);
  __bf16* xhln  = (__bf16*)alloc((size_t)1048576 * 2);
  float*  xhbn  = (float*) alloc((size_t)1048576 * 4);
  float*  xi    = (float*) alloc((size_t)2097152 * 4);
  float*  zbuf  = (float*) alloc((size_t)2097152 * 4);
  __bf16* xc    = (__bf16*)alloc((size_t)2097152 * 2);
  float*  xdbl  = (float*) alloc((size_t)3145728 * 4);
  float*  ysbuf = (float*) alloc((size_t)8388608 * 4);
  __bf16* gbuf  = (__bf16*)alloc((size_t)2097152 * 2);
  float*  hs    = (float*) alloc((size_t)1048576 * 4);
  __bf16* t2    = (__bf16*)alloc((size_t)1048576 * 2);

  __bf16* wb_pw1  = wbf;
  __bf16* wb_pw2  = wbf + 4096;
  __bf16* wb_inp  = wbf + 8192;
  __bf16* wb_outp = wbf + 24576;
  __bf16* wb_xp   = wbf + 32768;

  prep_weights  <<<224,  256, 0, stream>>>(pw1_w, pw2_w, in_proj_w, out_proj_w, x_proj_w, wbf);
  score_max     <<<256,  256, 0, stream>>>(x, score0);
  score_mlp     <<<4,     64, 0, stream>>>(score0, lin1_w, lin1_b, lin2_w, lin2_b, score);
  dwconv1_kernel<<<4096, 256, 0, stream>>>(x, dw1_w, dw1_b, t1);
  pw1_gemm      <<<128,  256, 0, stream>>>(t1, wb_pw1, pw1_b, hbuf);
  bstat_partial <<<1024, 256, 0, stream>>>(hbuf, psum, psq);
  bstat_final   <<<4,    256, 0, stream>>>(psum, psq, stats);
  bn_ln_kernel  <<<2048, 256, 0, stream>>>(hbuf, stats, ln_g, ln_b, xhbn, xhln);
  inproj_gemm   <<<512,  256, 0, stream>>>(xhln, wb_inp, in_proj_b, xi, zbuf);
  conv128_kernel<<<8192, 256, 0, stream>>>(xi, conv_w, conv_b, xc);
  xproj_gemm    <<<512,  256, 0, stream>>>(xc, wb_xp, xdbl);
  scan_kernel   <<<16,   128, 0, stream>>>(xdbl, xc, dt_proj_w, dt_proj_b, A_log, Ds, ysbuf);
  combine_kernel<<<2048, 256, 0, stream>>>(ysbuf, zbuf, out_norm_g, out_norm_b, gbuf);
  outproj_gemm  <<<128,  256, 0, stream>>>(gbuf, wb_outp, out_proj_b, xhbn, score, hs);
  dwconv2_kernel<<<4096, 256, 0, stream>>>(hs, dw2_w, dw2_b, t2);
  pw2_gemm      <<<128,  256, 0, stream>>>(t2, wb_pw2, pw2_b, x, out);
}
